// GATModel_39505109188898
// MI455X (gfx1250) — matrix operand order repaired and checked by reading, not compiled, not run on hardware
//
#include <hip/hip_runtime.h>
#include <hip/hip_bf16.h>

typedef __attribute__((ext_vector_type(16))) _Float16 v16h;
typedef __attribute__((ext_vector_type(8)))  _Float16 v8h;
typedef __attribute__((ext_vector_type(8)))  float    v8f;

#define IN_DIM 256
#define HID1   128   // HEADS*HID
#define HEADS  4
#define HIDC   32
#define OUTD   40
#define OUTPAD 48
#define NEG_SLOPE 0.2f

// ---------------------------------------------------------------------------
// float atomic max via sign-split integer atomics (monotonic float bit trick)
// ---------------------------------------------------------------------------
static __device__ __forceinline__ void atomicMaxF(float* addr, float val) {
  if (val >= 0.0f) atomicMax((int*)addr, __float_as_int(val));
  else             atomicMin((unsigned int*)addr, __float_as_uint(val));
}

// ---------------------------------------------------------------------------
// elementwise / staging kernels
// ---------------------------------------------------------------------------
__global__ void cast_pad_f16(const float* __restrict__ in, _Float16* __restrict__ out,
                             int rows, int cols, int rowsPad) {
  size_t i = (size_t)blockIdx.x * blockDim.x + threadIdx.x;
  size_t tot = (size_t)rowsPad * cols;
  if (i >= tot) return;
  int r = (int)(i / cols);
  out[i] = (r < rows) ? (_Float16)in[i] : (_Float16)0.0f;
}

// W: [K, C] row-major  ->  WT: [Cpad, K] row-major (zero-padded rows)
__global__ void transpose_f16(const float* __restrict__ W, _Float16* __restrict__ WT,
                              int K, int C, int Cpad) {
  size_t i = (size_t)blockIdx.x * blockDim.x + threadIdx.x;
  size_t tot = (size_t)Cpad * K;
  if (i >= tot) return;
  int n = (int)(i / K);
  int k = (int)(i % K);
  WT[i] = (n < C) ? (_Float16)W[(size_t)k * C + n] : (_Float16)0.0f;
}

__global__ void fill_f32(float* __restrict__ p, float v, size_t n) {
  size_t i = (size_t)blockIdx.x * blockDim.x + threadIdx.x;
  if (i < n) p[i] = v;
}

// ---------------------------------------------------------------------------
// WMMA GEMM: C[M,N] f32 = A[M,K] f16 (row-major) x B, with B given as
// BT[Npad,K] f16 (row-major transpose). One wave per 16x16 C tile.
// Fragment layouts per CDNA5 ISA §7.12.2 -> all per-lane data is
// contiguous 16-byte runs, so loads lower to global_load_b128 clauses.
// ---------------------------------------------------------------------------
__global__ void __launch_bounds__(32) wmma_gemm_f16f32(
    const _Float16* __restrict__ A, const _Float16* __restrict__ BT,
    float* __restrict__ C, int K, int ldc, int nmax) {
  const int lane = threadIdx.x;
  const int mt = blockIdx.x, nt = blockIdx.y;
  const _Float16* arow = A  + (size_t)(mt * 16 + (lane & 15)) * K;
  const _Float16* brow = BT + (size_t)(nt * 16 + (lane & 15)) * K;
  const int abase = (lane < 16) ? 0 : 8;
  const int bbase = (lane < 16) ? 0 : 8;
  v8f acc = {0.f, 0.f, 0.f, 0.f, 0.f, 0.f, 0.f, 0.f};
  for (int k0 = 0; k0 < K; k0 += 32) {
    union U { v16h v; v8h h[2]; } a, b;
    a.h[0] = *(const v8h*)(arow + k0 + abase);
    a.h[1] = *(const v8h*)(arow + k0 + 16 + abase);
    b.h[0] = *(const v8h*)(brow + k0 + bbase);
    b.h[1] = *(const v8h*)(brow + k0 + 16 + bbase);
    // 8 args: (neg_a, A, neg_b, B, c_mod, C, reuse_a, reuse_b)
    acc = __builtin_amdgcn_wmma_f32_16x16x32_f16(false, a.v, false, b.v,
                                                 (short)0, acc, false, false);
  }
  const int n = nt * 16 + (lane & 15);
  if (n < nmax) {
    const int mbase = mt * 16 + ((lane < 16) ? 0 : 8);
#pragma unroll
    for (int r = 0; r < 8; ++r) C[(size_t)(mbase + r) * ldc + n] = acc[r];
  }
}

// ---------------------------------------------------------------------------
// attention coefficients: as[n,h] = sum_c h[n,h,c]*a_src[h,c] (same for ad)
// ---------------------------------------------------------------------------
__global__ void alpha_kernel(const float* __restrict__ h, const float* __restrict__ a_src,
                             const float* __restrict__ a_dst, float* __restrict__ as,
                             float* __restrict__ ad, int Nn, int H, int ch) {
  int i = blockIdx.x * blockDim.x + threadIdx.x;
  if (i >= Nn * H) return;
  int node = i / H, hd = i % H;
  const float* hp = h + (size_t)node * H * ch + (size_t)hd * ch;
  const float* sp = a_src + (size_t)hd * ch;
  const float* dp = a_dst + (size_t)hd * ch;
  float s = 0.f, d = 0.f;
  for (int c = 0; c < ch; ++c) { float v = hp[c]; s += v * sp[c]; d += v * dp[c]; }
  as[i] = s;
  ad[i] = d;
}

// ---------------------------------------------------------------------------
// edge passes (edges E real + Nn self-loops); H compile-time -> shift math
// ---------------------------------------------------------------------------
static __device__ __forceinline__ void edge_ends(const int* __restrict__ ei, int E,
                                                 int eidx, int& s, int& d) {
  if (eidx < E) { s = ei[eidx]; d = ei[(size_t)E + eidx]; } else { s = d = eidx - E; }
}

template <int H>
__global__ void edge_score(const int* __restrict__ ei, int E, int Nn,
                           const float* __restrict__ as, const float* __restrict__ ad,
                           float* __restrict__ e, float* __restrict__ m) {
  size_t i = (size_t)blockIdx.x * blockDim.x + threadIdx.x;
  size_t tot = (size_t)(E + Nn) * H;
  if (i >= tot) return;
  int eidx = (int)(i / H), hd = (int)(i % H);
  int s, d;
  edge_ends(ei, E, eidx, s, d);
  float v = as[(size_t)s * H + hd] + ad[(size_t)d * H + hd];
  v = v > 0.f ? v : NEG_SLOPE * v;
  e[i] = v;
  atomicMaxF(&m[(size_t)d * H + hd], v);
}

template <int H>
__global__ void edge_exp(const int* __restrict__ ei, int E, int Nn,
                         float* __restrict__ e, const float* __restrict__ m,
                         float* __restrict__ den) {
  size_t i = (size_t)blockIdx.x * blockDim.x + threadIdx.x;
  size_t tot = (size_t)(E + Nn) * H;
  if (i >= tot) return;
  int eidx = (int)(i / H), hd = (int)(i % H);
  int d = (eidx < E) ? ei[(size_t)E + eidx] : (eidx - E);
  float t = __expf(e[i] - m[(size_t)d * H + hd]);
  e[i] = t;
  atomicAdd(&den[(size_t)d * H + hd], t);
}

// Vectorized scatter: 4 channels per thread -> one global_load_b128 gather of
// h[src], one alpha, 4 global_atomic_add_f32. hd is uniform per 4-group since
// ch (32 / 40) is a multiple of 4; row strides (128,40 floats) keep 16B align.
template <int H, int C>
__global__ void edge_aggr(const int* __restrict__ ei, int E, int Nn,
                          const float* __restrict__ h, const float* __restrict__ e,
                          const float* __restrict__ den, float* __restrict__ out) {
  constexpr int G  = C / 4;    // 4-channel groups per edge
  constexpr int ch = C / H;    // channels per head
  size_t i = (size_t)blockIdx.x * blockDim.x + threadIdx.x;
  size_t tot = (size_t)(E + Nn) * G;
  if (i >= tot) return;
  int eidx = (int)(i / G), g = (int)(i % G);
  int c = g * 4;
  int hd = c / ch;
  int s, d;
  edge_ends(ei, E, eidx, s, d);
  float alpha = e[(size_t)eidx * H + hd] / (den[(size_t)d * H + hd] + 1e-16f);
  const float4 hv = *(const float4*)(h + (size_t)s * C + c);
  float* op = out + (size_t)d * C + c;
  atomicAdd(op + 0, hv.x * alpha);
  atomicAdd(op + 1, hv.y * alpha);
  atomicAdd(op + 2, hv.z * alpha);
  atomicAdd(op + 3, hv.w * alpha);
}

// out1 + b1, ReLU, cast to f16 (zero padded rows) for next GEMM
__global__ void bias_relu_cast(const float* __restrict__ in, const float* __restrict__ b,
                               _Float16* __restrict__ out, int rows, int cols, int rowsPad) {
  size_t i = (size_t)blockIdx.x * blockDim.x + threadIdx.x;
  size_t tot = (size_t)rowsPad * cols;
  if (i >= tot) return;
  int r = (int)(i / cols), c = (int)(i % cols);
  float v = 0.f;
  if (r < rows) { v = in[i] + b[c]; v = v > 0.f ? v : 0.f; }
  out[i] = (_Float16)v;
}

// out[n,:] = log_softmax(in[n,:] + b2)
__global__ void bias_logsoftmax(const float* __restrict__ in, const float* __restrict__ b,
                                float* __restrict__ out, int Nn) {
  int node = blockIdx.x * blockDim.x + threadIdx.x;
  if (node >= Nn) return;
  float v[OUTD];
  float mx = -3.0e38f;
  for (int c = 0; c < OUTD; ++c) {
    v[c] = in[(size_t)node * OUTD + c] + b[c];
    mx = fmaxf(mx, v[c]);
  }
  float s = 0.f;
  for (int c = 0; c < OUTD; ++c) s += __expf(v[c] - mx);
  float lse = mx + logf(s);
  for (int c = 0; c < OUTD; ++c) out[(size_t)node * OUTD + c] = v[c] - lse;
}

// ---------------------------------------------------------------------------
extern "C" void kernel_launch(void* const* d_in, const int* in_sizes, int n_in,
                              void* d_out, int out_size, void* d_ws, size_t ws_size,
                              hipStream_t stream) {
  (void)n_in; (void)out_size; (void)ws_size;
  const float* x      = (const float*)d_in[0];
  const int*   ei     = (const int*)d_in[1];
  const float* W1     = (const float*)d_in[2];
  const float* a_src1 = (const float*)d_in[3];
  const float* a_dst1 = (const float*)d_in[4];
  const float* b1     = (const float*)d_in[5];
  const float* W2     = (const float*)d_in[6];
  const float* a_src2 = (const float*)d_in[7];
  const float* a_dst2 = (const float*)d_in[8];
  const float* b2     = (const float*)d_in[9];
  float* out = (float*)d_out;

  const int Nn   = in_sizes[0] / IN_DIM;
  const int E    = in_sizes[1] / 2;
  const int Mpad = (Nn + 15) & ~15;
  const int Etot = E + Nn;

  // ---- carve workspace (256B aligned), with explicit lifetime-based reuse
  char* p = (char*)d_ws;
  auto alloc = [&](size_t bytes) -> char* {
    char* r = p;
    p += (bytes + 255) & ~(size_t)255;
    return r;
  };
  // region A: x_h (layer-1 GEMM A); after gemm1 reused for hh (f16) + e2
  size_t offE2 = ((size_t)Mpad * HID1 * sizeof(_Float16) + 255) & ~(size_t)255;
  size_t szA1  = (size_t)Mpad * IN_DIM * sizeof(_Float16);
  size_t szA2  = offE2 + (size_t)Etot * sizeof(float);
  char* regA = alloc(szA1 > szA2 ? szA1 : szA2);
  _Float16* xh = (_Float16*)regA;
  _Float16* hh = (_Float16*)regA;                 // reuse after gemm1 consumed
  float*    e2 = (float*)(regA + offE2);
  // region C: h1 (f32); after layer-1 aggregation reused for h2 + out2
  size_t offO2 = ((size_t)Mpad * OUTD * sizeof(float) + 255) & ~(size_t)255;
  size_t szC1  = (size_t)Mpad * HID1 * sizeof(float);
  size_t szC2  = offO2 + (size_t)Nn * OUTD * sizeof(float);
  char* regC = alloc(szC1 > szC2 ? szC1 : szC2);
  float* h1   = (float*)regC;
  float* h2   = (float*)regC;                     // reuse after h1 dead
  float* out2 = (float*)(regC + offO2);

  float*    out1 = (float*)alloc((size_t)Mpad * HID1 * sizeof(float));
  float*    e1   = (float*)alloc((size_t)Etot * HEADS * sizeof(float));
  _Float16* w1t  = (_Float16*)alloc((size_t)HID1 * IN_DIM * sizeof(_Float16));
  _Float16* w2t  = (_Float16*)alloc((size_t)OUTPAD * HID1 * sizeof(_Float16));
  float* as1  = (float*)alloc((size_t)Nn * HEADS * sizeof(float));
  float* ad1  = (float*)alloc((size_t)Nn * HEADS * sizeof(float));
  float* m1   = (float*)alloc((size_t)Nn * HEADS * sizeof(float));
  float* den1 = (float*)alloc((size_t)Nn * HEADS * sizeof(float));
  float* as2  = (float*)alloc((size_t)Nn * sizeof(float));
  float* ad2  = (float*)alloc((size_t)Nn * sizeof(float));
  float* m2   = (float*)alloc((size_t)Nn * sizeof(float));
  float* den2 = (float*)alloc((size_t)Nn * sizeof(float));

  auto nb = [](size_t tot) { return (unsigned)((tot + 255) / 256); };
  size_t t;

  // ---- stage operands as f16 (A row-major, B transposed row-major)
  t = (size_t)Mpad * IN_DIM;
  cast_pad_f16<<<nb(t), 256, 0, stream>>>(x, xh, Nn, IN_DIM, Mpad);
  t = (size_t)HID1 * IN_DIM;
  transpose_f16<<<nb(t), 256, 0, stream>>>(W1, w1t, IN_DIM, HID1, HID1);
  t = (size_t)OUTPAD * HID1;
  transpose_f16<<<nb(t), 256, 0, stream>>>(W2, w2t, HID1, OUTD, OUTPAD);

  // ---- layer 1: h1 = x @ W1  (WMMA f16 -> f32)
  dim3 g1(Mpad / 16, HID1 / 16);
  wmma_gemm_f16f32<<<g1, 32, 0, stream>>>(xh, w1t, h1, IN_DIM, HID1, HID1);

  alpha_kernel<<<nb((size_t)Nn * HEADS), 256, 0, stream>>>(h1, a_src1, a_dst1,
                                                           as1, ad1, Nn, HEADS, HIDC);
  fill_f32<<<nb((size_t)Nn * HEADS), 256, 0, stream>>>(m1, -3.0e38f, (size_t)Nn * HEADS);
  fill_f32<<<nb((size_t)Nn * HEADS), 256, 0, stream>>>(den1, 0.f, (size_t)Nn * HEADS);
  fill_f32<<<nb((size_t)Nn * HID1), 256, 0, stream>>>(out1, 0.f, (size_t)Nn * HID1);

  t = (size_t)Etot * HEADS;
  edge_score<HEADS><<<nb(t), 256, 0, stream>>>(ei, E, Nn, as1, ad1, e1, m1);
  edge_exp<HEADS><<<nb(t), 256, 0, stream>>>(ei, E, Nn, e1, m1, den1);
  t = (size_t)Etot * (HID1 / 4);
  edge_aggr<HEADS, HID1><<<nb(t), 256, 0, stream>>>(ei, E, Nn, h1, e1, den1, out1);

  // ---- inter-layer: +b1, ReLU, cast to f16 (h1/x_h regions now reusable)
  t = (size_t)Mpad * HID1;
  bias_relu_cast<<<nb(t), 256, 0, stream>>>(out1, b1, hh, Nn, HID1, Mpad);

  // ---- layer 2: h2 = relu(out1+b1) @ W2 (WMMA), N padded 40 -> 48
  dim3 g2(Mpad / 16, OUTPAD / 16);
  wmma_gemm_f16f32<<<g2, 32, 0, stream>>>(hh, w2t, h2, HID1, OUTD, OUTD);

  alpha_kernel<<<nb((size_t)Nn), 256, 0, stream>>>(h2, a_src2, a_dst2, as2, ad2, Nn, 1, OUTD);
  fill_f32<<<nb((size_t)Nn), 256, 0, stream>>>(m2, -3.0e38f, (size_t)Nn);
  fill_f32<<<nb((size_t)Nn), 256, 0, stream>>>(den2, 0.f, (size_t)Nn);
  fill_f32<<<nb((size_t)Nn * OUTD), 256, 0, stream>>>(out2, 0.f, (size_t)Nn * OUTD);

  t = (size_t)Etot;
  edge_score<1><<<nb(t), 256, 0, stream>>>(ei, E, Nn, as2, ad2, e2, m2);
  edge_exp<1><<<nb(t), 256, 0, stream>>>(ei, E, Nn, e2, m2, den2);
  t = (size_t)Etot * (OUTD / 4);
  edge_aggr<1, OUTD><<<nb(t), 256, 0, stream>>>(ei, E, Nn, h2, e2, den2, out2);

  // ---- +b2 and log_softmax
  bias_logsoftmax<<<nb((size_t)Nn), 256, 0, stream>>>(out2, b2, out, Nn);
}
